// RelationalCrossAttention_61993557950539
// MI455X (gfx1250) — matrix operand order repaired and checked by reading, not compiled, not run on hardware
//
#include <hip/hip_runtime.h>
#include <cstdint>
#include <cstddef>

// ---------------------------------------------------------------------------
// CDNA5 (gfx1250) wave32 WMMA bf16 GEMM pipeline for relational cross-attn.
// D = A (MxK) * B^T (generic strides), f32 accumulate via
// v_wmma_f32_16x16x32_bf16.  Double-buffered LDS staging; bf16 tiles staged
// with GLOBAL_LOAD_ASYNC_TO_LDS_B128 (ASYNCcnt) when the builtin exists.
// Fragment layouts per CDNA5 ISA §7.12.2.
// ---------------------------------------------------------------------------

#if __has_builtin(__builtin_amdgcn_global_load_async_to_lds_b128)
#define HAVE_ASYNC_LDS 1
#else
#define HAVE_ASYNC_LDS 0
#endif

#if __has_builtin(__builtin_amdgcn_s_wait_asynccnt)
#define WAIT_ASYNC(N) __builtin_amdgcn_s_wait_asynccnt((unsigned short)(N))
#else
#define WAIT_ASYNC(N) asm volatile("s_wait_asynccnt %0" :: "i"(N) : "memory")
#endif

typedef __bf16 bf16_t;
typedef __attribute__((ext_vector_type(16))) __bf16 v16bf;
typedef __attribute__((ext_vector_type(8)))  float  v8f;
// GCC-style int4 vector: matches the async-LDS builtin's parameter type
// ('__attribute__((__vector_size__(4 * sizeof(int)))) int *').
typedef int v4i __attribute__((vector_size(16)));
#define AS1_V4I(p) ((__attribute__((address_space(1))) v4i*)(p))
#define AS3_V4I(p) ((__attribute__((address_space(3))) v4i*)(p))

union FragAB { uint4 q[2]; v16bf v; };
union AccU   { v8f v; float f[8]; };
union F4U    { float4 v; float f[4]; };
union H8U    { uint4 v; bf16_t h[8]; };
union H4U    { uint2 u; bf16_t h[4]; };

__global__ void cvt_f32_to_bf16(const float* __restrict__ in,
                                bf16_t* __restrict__ out, int n)
{
    int i = blockIdx.x * blockDim.x + threadIdx.x;
    if (i < n) out[i] = (bf16_t)in[i];
}

// EPI: 0 = store f32, 1 = store bf16, 2 = relu(acc*scale) -> f32
template<int BM, int BN, int WM, int WN, bool AF32, bool BKCONT, int EPI>
__global__ __launch_bounds__(WM*WN*32)
void gemm_wmma_bf16(const void* __restrict__ Ap, const bf16_t* __restrict__ Bp,
                    void* __restrict__ Cp,
                    int K, int ldaM, int ldbK, int ldbN, int ldc,
                    long sA1, long sA2, long sB1, long sB2, long sC1, long sC2,
                    int Z2, float scale)
{
    constexpr int BLOCK = WM*WN*32;
    constexpr int KB  = 32;                 // K per WMMA step (bf16 16x16x32)
    constexpr int PAD = 8;                  // bf16 elems of LDS padding
    constexpr int TM  = BM/(16*WM);
    constexpr int TN  = BN/(16*WN);
    constexpr int EA  = BM*KB/BLOCK;        // A elems staged per thread
    constexpr int EB  = BN*KB/BLOCK;        // B elems staged per thread
    // async ops issued per stage per thread (only bf16 vector paths)
    constexpr int XS  = HAVE_ASYNC_LDS ? ((AF32 ? 0 : EA/8) + (BKCONT ? EB/8 : 0)) : 0;
    static_assert(TM >= 1 && TN >= 1, "tiling");
    static_assert(AF32 ? (EA % 4 == 0) : (EA % 8 == 0), "A vector width");
    static_assert(EB % 8 == 0, "B vector width");

    __shared__ bf16_t As[2][BM][KB+PAD];
    __shared__ bf16_t Bs[2][BN][KB+PAD];

    const int tid   = threadIdx.x;
    const int lane  = tid & 31;
    const int wave  = tid >> 5;
    const int waveM = wave / WN;
    const int waveN = wave % WN;

    const int z  = blockIdx.z;
    const int z1 = z / Z2;
    const int z2 = z % Z2;
    const size_t offA = (size_t)sA1*z1 + (size_t)sA2*z2;
    const size_t offB = (size_t)sB1*z1 + (size_t)sB2*z2;
    const size_t offC = (size_t)sC1*z1 + (size_t)sC2*z2;

    const float*  Af = (const float*)Ap;
    const bf16_t* Ab = (const bf16_t*)Ap;

    const int m0 = blockIdx.y * BM;
    const int n0 = blockIdx.x * BN;

    const int laneM = lane & 15;
    const int hi    = lane >> 4;

    v8f acc[TM][TN] = {};

    // ---- tile staging: global -> LDS[sel] for K-slice kk -------------------
    auto stageTile = [&](int kk, int sel) {
        if (AF32) {                       // f32 source, convert + packed store
            #pragma unroll
            for (int e = 0; e < EA; e += 4) {
                int flat = tid*EA + e;
                int r = flat >> 5, c = flat & 31;
                const float* src = Af + offA + (size_t)(m0 + r)*ldaM + (kk + c);
                if (e == 0 && kk + KB < K)
                    __builtin_prefetch((const void*)(src + KB), 0, 1);
                F4U f; f.v = *(const float4*)src;
                H4U h;
                h.h[0] = (bf16_t)f.f[0];
                h.h[1] = (bf16_t)f.f[1];
                h.h[2] = (bf16_t)f.f[2];
                h.h[3] = (bf16_t)f.f[3];
                *(uint2*)&As[sel][r][c] = h.u;
            }
        } else {                          // bf16 source: async direct-to-LDS
            #pragma unroll
            for (int e = 0; e < EA; e += 8) {
                int flat = tid*EA + e;
                int r = flat >> 5, c = flat & 31;
                const bf16_t* src = Ab + offA + (size_t)(m0 + r)*ldaM + (kk + c);
#if HAVE_ASYNC_LDS
                __builtin_amdgcn_global_load_async_to_lds_b128(
                    AS1_V4I(src), AS3_V4I(&As[sel][r][c]), 0, 0);
#else
                *(uint4*)&As[sel][r][c] = *(const uint4*)src;
#endif
            }
        }
        if (BKCONT) {     // B^T(k,n) = Bp[n*ldbN + k], contiguous in k
            #pragma unroll
            for (int e = 0; e < EB; e += 8) {
                int flat = tid*EB + e;
                int n = flat >> 5, c = flat & 31;
                const bf16_t* src = Bp + offB + (size_t)(n0 + n)*ldbN + (kk + c);
#if HAVE_ASYNC_LDS
                __builtin_amdgcn_global_load_async_to_lds_b128(
                    AS1_V4I(src), AS3_V4I(&Bs[sel][n][c]), 0, 0);
#else
                *(uint4*)&Bs[sel][n][c] = *(const uint4*)src;
#endif
            }
        } else {          // B^T(k,n) = Bp[k*ldbK + n], contiguous in n: scatter
            #pragma unroll
            for (int e = 0; e < EB; e += 8) {
                int flat = tid*EB + e;
                int kk2 = flat / BN, nn = flat % BN;
                const bf16_t* src = Bp + offB + (size_t)(kk + kk2)*ldbK + (n0 + nn);
                if (e == 0 && kk + KB < K)
                    __builtin_prefetch((const void*)(src + (size_t)KB*ldbK), 0, 1);
                H8U u; u.v = *(const uint4*)src;
                #pragma unroll
                for (int ii = 0; ii < 8; ++ii) Bs[sel][nn+ii][kk2] = u.h[ii];
            }
        }
    };

    stageTile(0, 0);
    int buf = 0;

    for (int k0 = 0; k0 < K; k0 += KB) {
        // all waves finished reading buf^1 (their frag ds-loads drained
        // in the previous iteration, before they reach this barrier)
        __syncthreads();
        if (k0 + KB < K) {
            stageTile(k0 + KB, buf ^ 1);   // overlap: next tile in flight
            if (XS > 0) WAIT_ASYNC(XS);    // cur complete, next outstanding
        } else {
            if (XS > 0) WAIT_ASYNC(0);     // drain final stage
        }
        __syncthreads();                   // cur tile visible to all waves

        // -------- fragment loads, ISA §7.12.2 wave32 layouts --------------
        v16bf afrag[TM];
        #pragma unroll
        for (int tm = 0; tm < TM; ++tm) {
            int mloc = (waveM*TM + tm)*16 + laneM;
            FragAB fa;          // lane<16: K 0-7,16-23 ; lane>=16: K 8-15,24-31
            fa.q[0] = *(const uint4*)&As[buf][mloc][hi*8];
            fa.q[1] = *(const uint4*)&As[buf][mloc][hi*8 + 16];
            afrag[tm] = fa.v;
        }
        v16bf bfrag[TN];
        #pragma unroll
        for (int tn = 0; tn < TN; ++tn) {
            int nloc = (waveN*TN + tn)*16 + laneM;
            FragAB fb;          // lane<16: K 0-7,16-23 ; lane>=16: K 8-15,24-31
            fb.q[0] = *(const uint4*)&Bs[buf][nloc][hi*8];
            fb.q[1] = *(const uint4*)&Bs[buf][nloc][hi*8 + 16];
            bfrag[tn] = fb.v;
        }
        #pragma unroll
        for (int tm = 0; tm < TM; ++tm)
            #pragma unroll
            for (int tn = 0; tn < TN; ++tn)
                acc[tm][tn] = __builtin_amdgcn_wmma_f32_16x16x32_bf16(
                    /*neg_a=*/false, afrag[tm], /*neg_b=*/false, bfrag[tn],
                    /*c_mod=*/(short)0, acc[tm][tn],
                    /*reuse_a=*/false, /*reuse_b=*/false);

        buf ^= 1;
    }

    // ---------------- epilogue (C/D layout: VGPR r -> M=r / r+8) ----------
    float*  Cf = (float*)Cp;
    bf16_t* Cb = (bf16_t*)Cp;
    #pragma unroll
    for (int tm = 0; tm < TM; ++tm) {
        #pragma unroll
        for (int tn = 0; tn < TN; ++tn) {
            AccU cu; cu.v = acc[tm][tn];
            int mBase = m0 + (waveM*TM + tm)*16 + hi*8;
            int nIdx  = n0 + (waveN*TN + tn)*16 + laneM;
            #pragma unroll
            for (int r = 0; r < 8; ++r) {
                float val = cu.f[r];
                if (EPI == 2) val = fmaxf(val * scale, 0.0f);
                size_t idx = offC + (size_t)(mBase + r)*ldc + nIdx;
                if (EPI == 1) Cb[idx] = (bf16_t)val;
                else          Cf[idx] = val;
            }
        }
    }
}

extern "C" void kernel_launch(void* const* d_in, const int* in_sizes, int n_in,
                              void* d_out, int out_size, void* d_ws, size_t ws_size,
                              hipStream_t stream)
{
    constexpr int B = 16, S = 512, D = 512, H = 8, HD = 64;
    (void)in_sizes; (void)n_in; (void)out_size; (void)ws_size;

    const float* x   = (const float*)d_in[0];   // [B,S,D]
    const float* sym = (const float*)d_in[1];   // [S,S,D]
    const float* Wq  = (const float*)d_in[2];   // [D,D]
    const float* Wk  = (const float*)d_in[3];
    const float* Wv  = (const float*)d_in[4];
    const float* Wo  = (const float*)d_in[5];

    const size_t nX = (size_t)B*S*D;            // 4,194,304
    const size_t nW = (size_t)D*D;              //   262,144
    // workspace layout (bytes): bf16 copies/intermediates, ~290 MB total
    char* ws = (char*)d_ws;
    bf16_t* xb  = (bf16_t*)(ws);
    bf16_t* qb  = (bf16_t*)(ws + 2*nX);
    bf16_t* kb  = (bf16_t*)(ws + 4*nX);
    bf16_t* ab  = (bf16_t*)(ws + 6*nX);          // attn output pre-Wo
    bf16_t* wqb = (bf16_t*)(ws + 8*nX);
    bf16_t* wkb = (bf16_t*)(ws + 8*nX + 2*nW);
    bf16_t* wvb = (bf16_t*)(ws + 8*nX + 4*nW);
    bf16_t* wob = (bf16_t*)(ws + 8*nX + 6*nW);
    bf16_t* vb  = (bf16_t*)(ws + 8*nX + 8*nW);   // [S,S,D] bf16, 256 MB

    float* out_f    = (float*)d_out;             // output 0: [B,S,D]
    float* scores_f = out_f + nX;                // output 1: [B,H,S,S]

    // ---- bf16 conversions (x and weights; symbols/scores convert in-GEMM)
    cvt_f32_to_bf16<<<(int)((nX+255)/256),256,0,stream>>>(x,  xb,  (int)nX);
    cvt_f32_to_bf16<<<(int)((nW+255)/256),256,0,stream>>>(Wq, wqb, (int)nW);
    cvt_f32_to_bf16<<<(int)((nW+255)/256),256,0,stream>>>(Wk, wkb, (int)nW);
    cvt_f32_to_bf16<<<(int)((nW+255)/256),256,0,stream>>>(Wv, wvb, (int)nW);
    cvt_f32_to_bf16<<<(int)((nW+255)/256),256,0,stream>>>(Wo, wob, (int)nW);

    // ---- q = x @ Wq^T  ([8192x512]x[512x512] -> bf16)
    gemm_wmma_bf16<128,128,4,2,false,true,1>
        <<<dim3(D/128, (B*S)/128, 1), 256, 0, stream>>>(
        xb, wqb, qb, /*K*/D, /*ldaM*/D, /*ldbK*/1, /*ldbN*/D, /*ldc*/D,
        0,0,0,0,0,0, /*Z2*/1, 1.0f);
    // ---- k = x @ Wk^T
    gemm_wmma_bf16<128,128,4,2,false,true,1>
        <<<dim3(D/128, (B*S)/128, 1), 256, 0, stream>>>(
        xb, wkb, kb, D, D, 1, D, D, 0,0,0,0,0,0, 1, 1.0f);
    // ---- v = symbols @ Wv^T  (f32 A converted during LDS stage, bf16 out)
    gemm_wmma_bf16<128,128,4,2,true,true,1>
        <<<dim3(D/128, (S*S)/128, 1), 256, 0, stream>>>(
        sym, wvb, vb, D, D, 1, D, D, 0,0,0,0,0,0, 1, 1.0f);
    // ---- scores[b,h] = relu((q_h k_h^T)/8)  (batched over z=(b,h), f32 out)
    gemm_wmma_bf16<128,128,4,2,false,true,2>
        <<<dim3(S/128, S/128, B*H), 256, 0, stream>>>(
        qb, kb, scores_f, /*K*/HD, /*ldaM*/D, 1, /*ldbN*/D, /*ldc*/S,
        /*sA1*/(long)S*D, /*sA2*/HD, /*sB1*/(long)S*D, /*sB2*/HD,
        /*sC1*/(long)H*S*S, /*sC2*/(long)S*S, /*Z2*/H, 0.125f);
    // ---- attn[b,i,h,:] = sum_j scores[b,h,i,j] * v[i,j,h,:]
    //      batched over z=(i,h): [16 x 512] * [512 x 64], A=f32 scores,
    //      B strides: B^T(j,d)=v[(i*S+j)*D + h*64 + d]  (k-major)
    gemm_wmma_bf16<16,64,1,4,true,false,1>
        <<<dim3(1, 1, S*H), 128, 0, stream>>>(
        scores_f, vb, ab, /*K*/S, /*ldaM*/H*S*S, /*ldbK*/D, /*ldbN*/1,
        /*ldc*/S*D,
        /*sA1*/(long)S, /*sA2*/(long)S*S, /*sB1*/(long)S*D, /*sB2*/HD,
        /*sC1*/(long)D, /*sC2*/HD, /*Z2*/H, 1.0f);
    // ---- out = attn @ Wo^T  (f32 out into d_out[0])
    gemm_wmma_bf16<128,128,4,2,false,true,0>
        <<<dim3(D/128, (B*S)/128, 1), 256, 0, stream>>>(
        ab, wob, out_f, D, D, 1, D, D, 0,0,0,0,0,0, 1, 1.0f);
}
